// GateAttentionUnit_53042846106132
// MI455X (gfx1250) — compile-verified, matrix operand order checked by reading, not run on hardware
//
#include <hip/hip_runtime.h>
#include <hip/hip_bf16.h>

#define MAXSEQ 512
#define HID    1024
#define EE     2048
#define SS     128
#define FF     (2*EE + SS)     // 4224
#define BATCH  16
#define MTOT   (BATCH*MAXSEQ)  // 8192

typedef __attribute__((ext_vector_type(16))) _Float16 v16h;
typedef __attribute__((ext_vector_type(8)))  _Float16 v8h;
typedef __attribute__((ext_vector_type(8)))  float    v8f;

#define LDH 40   // LDS row stride in halfs: 32 data + 8 pad (80B rows, 16B aligned)

// gfx1250 async global->LDS copy path (ASYNCcnt), guarded so absence degrades to sync.
#if defined(__gfx1250__) && __has_builtin(__builtin_amdgcn_global_load_async_to_lds_b128) && \
    __has_builtin(__builtin_amdgcn_s_wait_asynccnt)
#define USE_ASYNC 1
typedef int v4i_t __attribute__((vector_size(16)));
typedef __attribute__((address_space(1))) v4i_t g_v4i;   // global (AS1)
typedef __attribute__((address_space(3))) v4i_t l_v4i;   // LDS (AS3)
#else
#define USE_ASYNC 0
#endif

__device__ __forceinline__ void cp16(const _Float16* g, _Float16* l) {
#if USE_ASYNC
  __builtin_amdgcn_global_load_async_to_lds_b128((g_v4i*)g, (l_v4i*)l, 0, 0);
#else
  *(v8h*)l = *(const v8h*)g;
#endif
}

__device__ __forceinline__ void wait_stage() {
#if USE_ASYNC
  __builtin_amdgcn_s_wait_asynccnt(0);
#endif
}

// Load a 16x32 f16 WMMA A/B fragment from an LDS tile (row-major, K contiguous).
// gfx1250 16-bit layout: lane group g = lane/16, row = lane%16;
// halfs [0..7] = K[8g..8g+7], halfs [8..15] = K[16+8g..23+8g]  -> two ds_load_b128.
__device__ __forceinline__ v16h ld_frag(const _Float16* tile, int row_base) {
  const int lane = threadIdx.x & 31;
  const _Float16* p = tile + (row_base + (lane & 15)) * LDH + ((lane >> 4) << 3);
  v8h lo = *(const v8h*)(p);
  v8h hi = *(const v8h*)(p + 16);
  v16h f;
#pragma unroll
  for (int i = 0; i < 8; ++i) { f[i] = lo[i]; f[i + 8] = hi[i]; }
  return f;
}

// ---------------- epilogue functors ----------------

struct EpiUV {
  const float* uvb; _Float16* u; _Float16* vt; float* basef;
  __device__ void operator()(int, int m, int n, float a) const {
    float f = a + uvb[n];
    float s = f / (1.0f + __expf(-f));
    if (n < EE) {
      u[(size_t)m * EE + n] = (_Float16)s;
    } else if (n < 2 * EE) {
      int b = m >> 9, r = m & 511, e = n - EE;
      vt[((size_t)b * EE + e) * MAXSEQ + r] = (_Float16)s;   // v^T: [b][e][m]
    } else {
      basef[(size_t)m * SS + (n - 2 * EE)] = s;
    }
  }
};

struct EpiQK {
  const float* wrel; _Float16* kern;
  __device__ void operator()(int b, int m, int n, float a) const {
    float t = a * (1.0f / (float)MAXSEQ) + wrel[511 + n - m];
    t = t > 0.0f ? t : 0.0f;
    kern[((size_t)b * MAXSEQ + m) * MAXSEQ + n] = (_Float16)(t * t);
  }
};

struct EpiKV {
  const _Float16* u; _Float16* gau;
  __device__ void operator()(int b, int m, int n, float a) const {
    size_t idx = ((size_t)(b * MAXSEQ + m)) * EE + n;
    gau[idx] = (_Float16)(a * (float)u[idx]);
  }
};

struct EpiOut {
  const float* ob; const float* x; float* out;
  __device__ void operator()(int, int m, int n, float a) const {
    size_t i = (size_t)m * HID + n;
    out[i] = a + ob[n] + x[i];
  }
};

// ---------------- generic NT WMMA GEMM ----------------
// C[M,N] = A[M,K] * B[N,K]^T, f16 in / f32 acc. Block tile 128x128, 8 waves of 32x64.
// Double-buffered LDS; tile t+1 staged (async on gfx1250) while tile t feeds WMMA.
template <typename Epi>
__global__ __launch_bounds__(256) void gemm_nt(const _Float16* __restrict__ A,
                                               const _Float16* __restrict__ B,
                                               int K, int lda, int ldb,
                                               long long sA, long long sB, Epi epi) {
  __shared__ __align__(16) _Float16 As[2][128 * LDH];
  __shared__ __align__(16) _Float16 Bs[2][128 * LDH];
  const int bz = blockIdx.z;
  A += (long long)bz * sA;
  B += (long long)bz * sB;
  const int bm = blockIdx.y * 128, bn = blockIdx.x * 128;
  const int tid = threadIdx.x;
  const int w  = tid >> 5;
  const int wm = (w >> 1) * 32;   // 4 waves along M
  const int wn = (w & 1) * 64;    // 2 waves along N

  // staging decomposition: 512 x 16B chunks per 128x32 tile, 2 per thread per matrix
  const int r0 = tid >> 2;          // 0..63
  const int c0 = (tid & 3) << 3;    // 0,8,16,24
  const _Float16* gA0 = A + (size_t)(bm + r0) * lda + c0;
  const _Float16* gB0 = B + (size_t)(bn + r0) * ldb + c0;
  const size_t lda64 = (size_t)lda * 64, ldb64 = (size_t)ldb * 64;
  const int lofs = r0 * LDH + c0;

  v8f acc[2][4];
#pragma unroll
  for (int i = 0; i < 2; ++i)
#pragma unroll
    for (int j = 0; j < 4; ++j)
#pragma unroll
      for (int e = 0; e < 8; ++e) acc[i][j][e] = 0.0f;

  const int nt = K >> 5;

  // prologue: stage tile 0 into buffer 0
  cp16(gA0, &As[0][lofs]);
  cp16(gB0, &Bs[0][lofs]);
  cp16(gA0 + lda64, &As[0][lofs + 64 * LDH]);
  cp16(gB0 + ldb64, &Bs[0][lofs + 64 * LDH]);

  int buf = 0;
  for (int t = 0; t < nt; ++t) {
    wait_stage();
    __syncthreads();

    if (t + 1 < nt) {  // stage next tile into the other buffer while computing
      const _Float16* ga = gA0 + ((t + 1) << 5);
      const _Float16* gb = gB0 + ((t + 1) << 5);
      _Float16* la = &As[buf ^ 1][lofs];
      _Float16* lb = &Bs[buf ^ 1][lofs];
      cp16(ga, la);
      cp16(gb, lb);
      cp16(ga + lda64, la + 64 * LDH);
      cp16(gb + ldb64, lb + 64 * LDH);
      __builtin_prefetch(ga + 32, 0, 1);  // pull tile t+2 toward L2
      __builtin_prefetch(gb + 32, 0, 1);
    }

    const _Float16* ta = As[buf];
    const _Float16* tb = Bs[buf];
    v16h af[2], bfr[4];
#pragma unroll
    for (int i = 0; i < 2; ++i) af[i] = ld_frag(ta, wm + 16 * i);
#pragma unroll
    for (int j = 0; j < 4; ++j) bfr[j] = ld_frag(tb, wn + 16 * j);
#pragma unroll
    for (int i = 0; i < 2; ++i)
#pragma unroll
      for (int j = 0; j < 4; ++j)
        acc[i][j] = __builtin_amdgcn_wmma_f32_16x16x32_f16(
            false, af[i], false, bfr[j], (short)0, acc[i][j], false, false);
    buf ^= 1;
  }

  // C/D layout: lane = N%16 column; VGPR e -> M = e + 8*(lane/16)
  const int lane = tid & 31;
  const int mo = (lane >> 4) * 8, no = lane & 15;
#pragma unroll
  for (int i = 0; i < 2; ++i)
#pragma unroll
    for (int j = 0; j < 4; ++j)
#pragma unroll
      for (int e = 0; e < 8; ++e)
        epi(bz, bm + wm + 16 * i + mo + e, bn + wn + 16 * j + no, acc[i][j][e]);
}

// ---------------- elementwise kernels ----------------

__global__ void cvt_kernel(const float* __restrict__ in, _Float16* __restrict__ out, int n) {
  int i = blockIdx.x * blockDim.x + threadIdx.x;
  int stride = gridDim.x * blockDim.x;
  for (; i < n; i += stride) out[i] = (_Float16)in[i];
}

__global__ __launch_bounds__(256) void ln_f16_kernel(const float* __restrict__ x,
                                                     const float* __restrict__ g,
                                                     const float* __restrict__ bta,
                                                     _Float16* __restrict__ xn) {
  __shared__ float rs_[256];
  __shared__ float rq_[256];
  const int row = blockIdx.x, tid = threadIdx.x;
  const float* xr = x + (size_t)row * HID;
  float v[4]; float s = 0.0f, q = 0.0f;
#pragma unroll
  for (int i = 0; i < 4; ++i) { v[i] = xr[tid + i * 256]; s += v[i]; q += v[i] * v[i]; }
  rs_[tid] = s; rq_[tid] = q; __syncthreads();
  for (int off = 128; off > 0; off >>= 1) {
    if (tid < off) { rs_[tid] += rs_[tid + off]; rq_[tid] += rq_[tid + off]; }
    __syncthreads();
  }
  const float mu  = rs_[0] * (1.0f / HID);
  const float var = rq_[0] * (1.0f / HID) - mu * mu;
  const float r   = rsqrtf(var + 1e-5f);
#pragma unroll
  for (int i = 0; i < 4; ++i) {
    int c = tid + i * 256;
    xn[(size_t)row * HID + c] = (_Float16)((v[i] - mu) * r * g[c] + bta[c]);
  }
}

__global__ __launch_bounds__(256) void rope_qk_kernel(const float* __restrict__ basef,
                                                      const float* __restrict__ gamma,
                                                      const float* __restrict__ beta,
                                                      _Float16* __restrict__ qh,
                                                      _Float16* __restrict__ kh) {
  const int t = threadIdx.x;
  const int row = blockIdx.x * 4 + (t >> 6);  // 0..8191
  const int j = t & 63;
  const int n = row & (MAXSEQ - 1);
  const float b1 = basef[(size_t)row * SS + j];
  const float b2 = basef[(size_t)row * SS + j + 64];
  const float freq = __powf(10000.0f, (float)j * (1.0f / 64.0f));
  float sn, cs;
  __sincosf((float)n * freq, &sn, &cs);
#pragma unroll
  for (int g = 0; g < 2; ++g) {
    const float t1 = b1 * gamma[g * SS + j]      + beta[g * SS + j];
    const float t2 = b2 * gamma[g * SS + j + 64] + beta[g * SS + j + 64];
    _Float16* dst = g ? kh : qh;
    dst[(size_t)row * SS + j]      = (_Float16)(t1 * cs - t2 * sn);
    dst[(size_t)row * SS + j + 64] = (_Float16)(t2 * cs + t1 * sn);
  }
}

// ---------------- host launcher ----------------

extern "C" void kernel_launch(void* const* d_in, const int* in_sizes, int n_in,
                              void* d_out, int out_size, void* d_ws, size_t ws_size,
                              hipStream_t stream) {
  (void)in_sizes; (void)n_in; (void)out_size; (void)ws_size;
  const float* x     = (const float*)d_in[0];
  const float* ln_w  = (const float*)d_in[1];
  const float* ln_b  = (const float*)d_in[2];
  const float* uv_w  = (const float*)d_in[3];
  const float* uv_b  = (const float*)d_in[4];
  const float* gamma = (const float*)d_in[5];
  const float* beta  = (const float*)d_in[6];
  const float* o_w   = (const float*)d_in[7];
  const float* o_b   = (const float*)d_in[8];
  const float* wrel  = (const float*)d_in[9];
  float* out = (float*)d_out;

  char* ws = (char*)d_ws;
  size_t off = 0;
  auto take = [&](size_t bytes) -> char* {
    char* p = ws + off;
    off = (off + bytes + 255) & ~(size_t)255;
    return p;
  };
  _Float16* xn_h   = (_Float16*)take((size_t)MTOT * HID * 2);
  _Float16* wuv_h  = (_Float16*)take((size_t)FF * HID * 2);
  _Float16* wo_h   = (_Float16*)take((size_t)HID * EE * 2);
  _Float16* u_h    = (_Float16*)take((size_t)MTOT * EE * 2);
  _Float16* vt_h   = (_Float16*)take((size_t)MTOT * EE * 2);
  float*    basef  = (float*)   take((size_t)MTOT * SS * 4);
  _Float16* q_h    = (_Float16*)take((size_t)MTOT * SS * 2);
  _Float16* k_h    = (_Float16*)take((size_t)MTOT * SS * 2);
  _Float16* kern_h = (_Float16*)take((size_t)BATCH * MAXSEQ * MAXSEQ * 2);
  _Float16* gau_h  = (_Float16*)take((size_t)MTOT * EE * 2);

  cvt_kernel<<<dim3(512), dim3(256), 0, stream>>>(uv_w, wuv_h, FF * HID);
  cvt_kernel<<<dim3(512), dim3(256), 0, stream>>>(o_w,  wo_h,  HID * EE);

  ln_f16_kernel<<<dim3(MTOT), dim3(256), 0, stream>>>(x, ln_w, ln_b, xn_h);

  // GEMM1: uv = silu(xn @ uv_w^T + uv_b); scatter u / v^T / base
  gemm_nt<<<dim3(FF / 128, MTOT / 128, 1), dim3(256), 0, stream>>>(
      xn_h, wuv_h, HID, HID, HID, 0LL, 0LL, EpiUV{uv_b, u_h, vt_h, basef});

  rope_qk_kernel<<<dim3(MTOT / 4), dim3(256), 0, stream>>>(basef, gamma, beta, q_h, k_h);

  // GEMM2 (batched): kernel = sq_relu(q @ k^T / 512 + rel_bias)
  gemm_nt<<<dim3(MAXSEQ / 128, MAXSEQ / 128, BATCH), dim3(256), 0, stream>>>(
      q_h, k_h, SS, SS, SS, (long long)MAXSEQ * SS, (long long)MAXSEQ * SS,
      EpiQK{wrel, kern_h});

  // GEMM3 (batched): gau = u * (kernel @ v)
  gemm_nt<<<dim3(EE / 128, MAXSEQ / 128, BATCH), dim3(256), 0, stream>>>(
      kern_h, vt_h, MAXSEQ, MAXSEQ, MAXSEQ, (long long)MAXSEQ * MAXSEQ,
      (long long)EE * MAXSEQ, EpiKV{u_h, gau_h});

  // GEMM4: out = gau @ o_w^T + o_b + x
  gemm_nt<<<dim3(HID / 128, MTOT / 128, 1), dim3(256), 0, stream>>>(
      gau_h, wo_h, EE, EE, EE, 0LL, 0LL, EpiOut{o_b, x, out});
}
